// LSTM_35124242547024
// MI455X (gfx1250) — matrix operand order repaired and checked by reading, not compiled, not run on hardware
//
#include <hip/hip_runtime.h>
#include <hip/hip_bf16.h>

typedef __attribute__((ext_vector_type(16))) _Float16 v16h;
typedef __attribute__((ext_vector_type(8)))  float    v8f;

#define B_  256
#define T_  512
#define D_  512
#define H_  64
#define G4  256   // 4*H

// ---------- small math helpers ----------
static __device__ __forceinline__ float fast_sig(float x) {
    return 1.f / (1.f + __expf(-x));
}
static __device__ __forceinline__ float fast_tanh(float x) {
    float e = __expf(-2.f * x);
    return (1.f - e) / (1.f + e);
}

// ---------- WMMA fragment loaders (wave32, 16x16x32 f16) ----------
// A matrix 16x32 (MxK), ld in halfs. lane&15 = M; lanes>=16 use K+8 offset.
static __device__ __forceinline__ v16h load_a_frag(const _Float16* As, int ld, int lane) {
    union { v16h h; unsigned u[8]; } f;
    const int m  = lane & 15;
    const int kb = (lane >> 4) << 3;   // 0 or 8
#pragma unroll
    for (int j = 0; j < 8; ++j) {
        const int kk = ((j >> 2) << 4) + kb + ((j & 3) << 1);
        f.u[j] = *(const unsigned*)(As + m * ld + kk);
    }
    return f.h;
}

// B matrix 32x16 (KxN), stored row-major K x ld; lane&15 = N, elements over K (same K order as A).
static __device__ __forceinline__ v16h load_b_frag(const _Float16* Bs, int ld, int n0, int lane) {
    union { v16h h; _Float16 e[16]; } f;
    const int n  = lane & 15;
    const int kb = (lane >> 4) << 3;
#pragma unroll
    for (int s = 0; s < 16; ++s) {
        const int k = ((s >> 3) << 4) + kb + (s & 7);
        f.e[s] = Bs[(size_t)k * ld + n0 + n];
    }
    return f.h;
}

// ---------- prep: transpose weights to f16, combine biases ----------
__global__ __launch_bounds__(256) void prep_kernel(const float* __restrict__ Wih,
                                                   const float* __restrict__ Whh,
                                                   const float* __restrict__ bih,
                                                   const float* __restrict__ bhh,
                                                   _Float16* __restrict__ WihT,
                                                   _Float16* __restrict__ WhhT,
                                                   float* __restrict__ bsum) {
    int i = blockIdx.x * 256 + threadIdx.x;
    if (i < G4 * D_) {                       // W_ih [256][512] -> WihT [512][256]
        int g = i >> 9, d = i & 511;
        WihT[d * G4 + g] = (_Float16)Wih[g * D_ + d];
    } else if (i < G4 * D_ + G4 * H_) {      // W_hh [256][64] -> WhhT [64][256]
        int j = i - G4 * D_;
        int g = j >> 6, k = j & 63;
        WhhT[k * G4 + g] = (_Float16)Whh[g * H_ + k];
    } else if (i < G4 * D_ + G4 * H_ + G4) { // bsum = b_ih + b_hh
        int j = i - G4 * D_ - G4 * H_;
        bsum[j] = bih[j] + bhh[j];
    }
}

// ---------- x_proj GEMM: [M=131072 x K=512] * [K x N=256], f16 WMMA ----------
// Block: 256 thr (8 waves). BM=64, BN=256 (full N), BK=32, double-buffered.
// B tile is DMA'd global->LDS via GLOBAL_LOAD_ASYNC_TO_LDS_B128 (ASYNCcnt).
__global__ __launch_bounds__(256) void xproj_gemm(const float* __restrict__ x,
                                                  const _Float16* __restrict__ WihT,
                                                  float* __restrict__ xproj) {
    __shared__ _Float16 As[2][64 * 32];   //  8 KB
    __shared__ _Float16 Bs[2][32 * G4];   // 32 KB
    const int tid   = threadIdx.x;
    const int lane  = tid & 31;
    const int wv    = tid >> 5;
    const int mi    = wv >> 1;
    const int nbase = (wv & 1) * 128;
    const size_t m0 = (size_t)blockIdx.x * 64;

    // staging coordinates
    const int arow = tid >> 2;           // A: 64 rows x 32 halfs, 4 thr/row (32B each)
    const int acg  = (tid & 3) * 8;
    const int brow = tid >> 3;           // B: 32 rows x 256 halfs, 8 thr/row (64B each)
    const int bcg  = (tid & 7) * 32;

    const float*    aSrc = x + (m0 + arow) * D_ + acg;
    const _Float16* bSrc = WihT + (size_t)brow * G4 + bcg;

    v8f acc[8];
#pragma unroll
    for (int i = 0; i < 8; ++i) acc[i] = (v8f){};

#define ISSUE_B(kt, buf)                                                          \
    {                                                                             \
        unsigned lds = (unsigned)(uintptr_t)(&Bs[buf][brow * G4 + bcg]);          \
        unsigned long long ga =                                                   \
            (unsigned long long)(uintptr_t)(bSrc + (size_t)(kt) * 32 * G4);       \
        asm volatile(                                                             \
            "global_load_async_to_lds_b128 %0, %1, off\n\t"                       \
            "global_load_async_to_lds_b128 %0, %1, off offset:16\n\t"             \
            "global_load_async_to_lds_b128 %0, %1, off offset:32\n\t"             \
            "global_load_async_to_lds_b128 %0, %1, off offset:48"                 \
            :: "v"(lds), "v"(ga) : "memory");                                     \
    }

#define STORE_A(buf, f0, f1)                                                      \
    {                                                                             \
        _Float16* dst = &As[buf][arow * 32 + acg];                                \
        dst[0] = (_Float16)f0.x; dst[1] = (_Float16)f0.y;                         \
        dst[2] = (_Float16)f0.z; dst[3] = (_Float16)f0.w;                         \
        dst[4] = (_Float16)f1.x; dst[5] = (_Float16)f1.y;                         \
        dst[6] = (_Float16)f1.z; dst[7] = (_Float16)f1.w;                         \
    }

    // prologue: stage kt=0 into buffer 0
    ISSUE_B(0, 0);
    {
        const float4* gp = (const float4*)aSrc;
        float4 f0 = gp[0], f1 = gp[1];
        STORE_A(0, f0, f1);
    }
    asm volatile("s_wait_asynccnt 0" ::: "memory");
    __syncthreads();

    const int NK = D_ / 32;
    for (int kt = 0; kt < NK; ++kt) {
        const int cur = kt & 1, nxt = cur ^ 1;
        float4 f0, f1;
        if (kt + 1 < NK) {
            ISSUE_B(kt + 1, nxt);                        // async DMA, overlaps WMMA
            const float4* gp = (const float4*)(aSrc + (kt + 1) * 32);
            f0 = gp[0]; f1 = gp[1];                      // fp32 loads in flight
        }

        v16h a = load_a_frag(&As[cur][mi * 16 * 32], 32, lane);
#pragma unroll
        for (int nt = 0; nt < 8; ++nt) {
            v16h b = load_b_frag(&Bs[cur][0], G4, nbase + nt * 16, lane);
            acc[nt] = __builtin_amdgcn_wmma_f32_16x16x32_f16(
                false, a, false, b, (short)0, acc[nt], false, false);
        }

        if (kt + 1 < NK) {
            STORE_A(nxt, f0, f1);                        // convert + LDS store
            asm volatile("s_wait_asynccnt 0" ::: "memory");
        }
        __syncthreads();
    }
#undef ISSUE_B
#undef STORE_A

    // write C: VGPR r -> M = r (+8 for upper lanes), N = lane&15
    const int mlo = (lane >> 4) << 3;
    const int nn  = lane & 15;
#pragma unroll
    for (int nt = 0; nt < 8; ++nt) {
        const int col = nbase + nt * 16 + nn;
#pragma unroll
        for (int r = 0; r < 8; ++r) {
            const size_t row = m0 + mi * 16 + r + mlo;
            xproj[row * G4 + col] = acc[nt][r];
        }
    }
}

// ---------- LSTM scan: 16 blocks x 16 batch rows, 512 sequential steps ----------
__global__ __launch_bounds__(256) void lstm_scan(const float* __restrict__ xproj,
                                                 const _Float16* __restrict__ WhhT,
                                                 const float* __restrict__ bsum,
                                                 float* __restrict__ pooled) {
    __shared__ _Float16 h16[16 * H_];   // h state as f16 (A operand)
    __shared__ float    gates[16 * G4]; // WMMA results
    const int tid  = threadIdx.x;
    const int lane = tid & 31;
    const int wv   = tid >> 5;
    const int bb   = blockIdx.x * 16;
    const int n0   = wv * 32;

    // W_hh^T fragments live in registers for the whole scan (K=64 -> 2 chunks)
    v16h Bf[2][2];
#pragma unroll
    for (int kc = 0; kc < 2; ++kc)
#pragma unroll
        for (int t2 = 0; t2 < 2; ++t2)
            Bf[kc][t2] = load_b_frag(WhhT + kc * 32 * G4, G4, n0 + t2 * 16, lane);

    for (int i = tid; i < 16 * H_; i += 256) h16[i] = (_Float16)0.f;

    // per-thread (b,h) ownership + hoisted biases and xproj row bases
    int   bOf[4], hOf[4];
    float bsv[4][4];
    size_t rBase[4];
#pragma unroll
    for (int j = 0; j < 4; ++j) {
        const int p = tid + j * 256;
        bOf[j] = p >> 6;
        hOf[j] = p & 63;
        rBase[j] = ((size_t)(bb + bOf[j]) * T_) * G4;
#pragma unroll
        for (int g = 0; g < 4; ++g) bsv[j][g] = bsum[g * 64 + hOf[j]];
    }
    float cst[4]  = {0.f, 0.f, 0.f, 0.f};
    float hsum[4] = {0.f, 0.f, 0.f, 0.f};
    __syncthreads();

    const int mlo = (lane >> 4) << 3;
    const int nn  = lane & 15;

    for (int t = 0; t < T_; ++t) {
        // issue this step's xproj gate loads early; they complete under the WMMAs
        float xp[4][4];
#pragma unroll
        for (int j = 0; j < 4; ++j) {
            const float* p = xproj + rBase[j] + (size_t)t * G4 + hOf[j];
            xp[j][0] = p[0]; xp[j][1] = p[64]; xp[j][2] = p[128]; xp[j][3] = p[192];
            if (t + 1 < T_) {                       // warm next step (global_prefetch_b8)
                __builtin_prefetch(p + G4, 0, 1);
                __builtin_prefetch(p + G4 + 128, 0, 1);
            }
        }

        v8f acc0 = (v8f){}, acc1 = (v8f){};
#pragma unroll
        for (int kc = 0; kc < 2; ++kc) {
            v16h a = load_a_frag(h16 + kc * 32, H_, lane);
            acc0 = __builtin_amdgcn_wmma_f32_16x16x32_f16(
                false, a, false, Bf[kc][0], (short)0, acc0, false, false);
            acc1 = __builtin_amdgcn_wmma_f32_16x16x32_f16(
                false, a, false, Bf[kc][1], (short)0, acc1, false, false);
        }
#pragma unroll
        for (int r = 0; r < 8; ++r) {
            const int m = r + mlo;
            gates[m * G4 + n0 + nn]      = acc0[r];
            gates[m * G4 + n0 + 16 + nn] = acc1[r];
        }
        __syncthreads();

        // elementwise LSTM cell: each thread owns 4 (b,h) pairs
#pragma unroll
        for (int j = 0; j < 4; ++j) {
            const int b = bOf[j], h = hOf[j];
            float iv = gates[b * G4 + h]       + xp[j][0] + bsv[j][0];
            float fv = gates[b * G4 + 64 + h]  + xp[j][1] + bsv[j][1];
            float gv = gates[b * G4 + 128 + h] + xp[j][2] + bsv[j][2];
            float ov = gates[b * G4 + 192 + h] + xp[j][3] + bsv[j][3];
            iv = fast_sig(iv); fv = fast_sig(fv); gv = fast_tanh(gv); ov = fast_sig(ov);
            cst[j] = fv * cst[j] + iv * gv;
            const float hv = ov * fast_tanh(cst[j]);
            hsum[j] += hv;
            h16[b * H_ + h] = (_Float16)hv;
        }
        __syncthreads();
    }

#pragma unroll
    for (int j = 0; j < 4; ++j)
        pooled[(size_t)(bb + bOf[j]) * H_ + hOf[j]] = hsum[j] * (1.f / (float)T_);
}

// ---------- head: relu(pooled@W1^T+b1)@W2^T+b2 -> softmax ----------
__global__ __launch_bounds__(256) void head_mlp(const float* __restrict__ pooled,
                                                const float* __restrict__ W1,
                                                const float* __restrict__ b1,
                                                const float* __restrict__ W2,
                                                const float* __restrict__ b2,
                                                float* __restrict__ out) {
    const int b = blockIdx.x * blockDim.x + threadIdx.x;
    if (b >= B_) return;
    float hbuf[32];
#pragma unroll
    for (int j = 0; j < 32; ++j) {
        float s = b1[j];
        for (int k = 0; k < H_; ++k) s += pooled[b * H_ + k] * W1[j * H_ + k];
        hbuf[j] = fmaxf(s, 0.f);
    }
    float lg[3];
#pragma unroll
    for (int j = 0; j < 3; ++j) {
        float s = b2[j];
#pragma unroll
        for (int k = 0; k < 32; ++k) s += hbuf[k] * W2[j * 32 + k];
        lg[j] = s;
    }
    float m = fmaxf(lg[0], fmaxf(lg[1], lg[2]));
    float e0 = __expf(lg[0] - m), e1 = __expf(lg[1] - m), e2 = __expf(lg[2] - m);
    float inv = 1.f / (e0 + e1 + e2);
    out[b * 3 + 0] = e0 * inv;
    out[b * 3 + 1] = e1 * inv;
    out[b * 3 + 2] = e2 * inv;
}

extern "C" void kernel_launch(void* const* d_in, const int* in_sizes, int n_in,
                              void* d_out, int out_size, void* d_ws, size_t ws_size,
                              hipStream_t stream) {
    const float* x    = (const float*)d_in[0];
    const float* Wih  = (const float*)d_in[1];
    const float* Whh  = (const float*)d_in[2];
    const float* bih  = (const float*)d_in[3];
    const float* bhh  = (const float*)d_in[4];
    const float* W1   = (const float*)d_in[5];
    const float* b1   = (const float*)d_in[6];
    const float* W2   = (const float*)d_in[7];
    const float* b2   = (const float*)d_in[8];
    float* out = (float*)d_out;

    // workspace partition
    char* w = (char*)d_ws;
    _Float16* WihT = (_Float16*)w;  w += (size_t)D_ * G4 * sizeof(_Float16);   // 256 KB
    _Float16* WhhT = (_Float16*)w;  w += (size_t)H_ * G4 * sizeof(_Float16);   // 32 KB
    float* bsum    = (float*)w;     w += (size_t)G4 * sizeof(float);           // 1 KB
    float* pooled  = (float*)w;     w += (size_t)B_ * H_ * sizeof(float);      // 64 KB
    w = (char*)(((uintptr_t)w + 255) & ~(uintptr_t)255);
    float* xproj   = (float*)w;     // 131072 * 256 * 4 = 128 MB

    // 1) weight transpose/convert + bias merge
    {
        int total = G4 * D_ + G4 * H_ + G4;
        prep_kernel<<<(total + 255) / 256, 256, 0, stream>>>(Wih, Whh, bih, bhh,
                                                             WihT, WhhT, bsum);
    }
    // 2) big input-projection GEMM (async-LDS double-buffered WMMA f16 -> f32)
    xproj_gemm<<<(B_ * T_) / 64, 256, 0, stream>>>(x, WihT, xproj);
    // 3) sequential LSTM scan with in-register W_hh fragments
    lstm_scan<<<B_ / 16, 256, 0, stream>>>(xproj, WhhT, bsum, pooled);
    // 4) tiny MLP head + softmax
    head_mlp<<<1, 256, 0, stream>>>(pooled, W1, b1, W2, b2, out);
}
